// MultiAttentionHead_27676769255690
// MI455X (gfx1250) — matrix operand order repaired and checked by reading, not compiled, not run on hardware
//
#include <hip/hip_runtime.h>
#include <hip/hip_bf16.h>

#define EMB   1024
#define HEADS 16
#define HDIM  64
#define BATCH 2
#define SEQ   2048

typedef __attribute__((ext_vector_type(16))) __bf16   v16bf;
typedef __attribute__((ext_vector_type(8)))  float    v8f;
typedef __attribute__((ext_vector_type(8)))  unsigned v8u;
typedef __attribute__((ext_vector_type(4)))  unsigned v4u;

// ---------- bf16 helpers (manual RNE) ----------
static __device__ __forceinline__ unsigned short f2bfu(float f) {
    unsigned u; __builtin_memcpy(&u, &f, 4);
    u = (u + 0x7FFFu + ((u >> 16) & 1u)) >> 16;
    return (unsigned short)u;
}
static __device__ __forceinline__ unsigned pk2(float lo, float hi) {
    return (unsigned)f2bfu(lo) | ((unsigned)f2bfu(hi) << 16);
}
static __device__ __forceinline__ v16bf frag_from2(v4u lo, v4u hi) {
    v8u u;
#pragma unroll
    for (int i = 0; i < 4; ++i) { u[i] = lo[i]; u[i + 4] = hi[i]; }
    return __builtin_bit_cast(v16bf, u);
}

// ---------- WMMA fragment loaders (two b128 loads each) ----------
static __device__ __forceinline__ v16bf load_a_bf16(const unsigned short* base, int stride, int lane) {
    const int row = lane & 15, half = lane >> 4;
    const unsigned short* p = base + (size_t)row * stride + 8 * half;
    return frag_from2(*(const v4u*)p, *(const v4u*)(p + 16));
}
static __device__ __forceinline__ v16bf load_b_col_bf16(const unsigned short* base, int stride, int lane) {
    const int col = lane & 15, half = lane >> 4;
    const unsigned short* p = base + (size_t)col * stride + 8 * half;
    return frag_from2(*(const v4u*)p, *(const v4u*)(p + 16));
}

static __device__ __forceinline__ v8f wmma_bf16(v16bf a, v16bf b, v8f c) {
    return __builtin_amdgcn_wmma_f32_16x16x32_bf16(false, a, false, b, (short)0, c, false, false);
}

// ---------- async LDS staging (CDNA5 ASYNCcnt path) ----------
static __device__ __forceinline__ void async_b128(unsigned lds, unsigned voff, const void* base) {
    asm volatile("global_load_async_to_lds_b128 %0, %1, %2"
                 :: "v"(lds), "v"(voff), "s"((unsigned long long)(uintptr_t)base) : "memory");
}
static __device__ __forceinline__ void wait_async0() { asm volatile("s_wait_asynccnt 0x0" ::: "memory"); }
static __device__ __forceinline__ void wait_async1() { asm volatile("s_wait_asynccnt 0x1" ::: "memory"); }
static __device__ __forceinline__ void wait_async2() { asm volatile("s_wait_asynccnt 0x2" ::: "memory"); }

// =====================================================================
// One-shot precision conversion kernels
// =====================================================================
__global__ __launch_bounds__(256) void cvt_kernel(const float* __restrict__ src,
                                                  unsigned short* __restrict__ dst, int n8) {
    const int i = blockIdx.x * 256 + threadIdx.x;
    if (i >= n8) return;
    const float4* s = (const float4*)src + (size_t)i * 2;
    const float4 a = s[0], b = s[1];
    v4u o; o[0] = pk2(a.x, a.y); o[1] = pk2(a.z, a.w); o[2] = pk2(b.x, b.y); o[3] = pk2(b.z, b.w);
    *(v4u*)(dst + (size_t)i * 8) = o;
}
// W [H,E,D] f32 -> Wt [H,D,E] bf16 (contraction dim made contiguous)
__global__ __launch_bounds__(256) void cvtT_kernel(const float* __restrict__ src,
                                                   unsigned short* __restrict__ dst) {
    const int t  = blockIdx.x * 256 + threadIdx.x;   // H*E*(D/2) threads
    const int d2 = t & (HDIM / 2 - 1);
    const int e  = (t >> 5) & (EMB - 1);
    const int h  = t >> 15;
    const float* s = src + ((size_t)h * EMB + e) * HDIM + 2 * d2;
    unsigned short* dT = dst + (size_t)h * HDIM * EMB + e;
    dT[(size_t)(2 * d2)     * EMB] = f2bfu(s[0]);
    dT[(size_t)(2 * d2 + 1) * EMB] = f2bfu(s[1]);
}

// =====================================================================
// Kernel 1: Q/K/V projections. 128x64 tile per block; weight chunks
// async-staged to DOUBLE-BUFFERED LDS so the next chunk's loads are in
// flight during this chunk's WMMAs. V stored TRANSPOSED [B,H,D,T].
// =====================================================================
__global__ __launch_bounds__(256) void proj_qkv_kernel(
    const unsigned short* __restrict__ xb, const unsigned short* __restrict__ wt,
    unsigned short* __restrict__ qkv)
{
    __shared__ unsigned short wbuf[2][64 * 32];   // Wt chunk [d 64][e 32], double-buffered

    const int tid = threadIdx.x;
    const int lane = tid & 31;
    const int wave = tid >> 5;
    const int h = blockIdx.y;
    const int which = blockIdx.z % 3;
    const int b = blockIdx.z / 3;
    const int row0 = blockIdx.x * 128 + wave * 16;

    const unsigned short* Wth = wt + ((size_t)which * HEADS + h) * HDIM * EMB;  // [D][E]
    const unsigned short* Xw  = xb + ((size_t)b * SEQ + row0) * EMB;

    const int vrow = tid >> 2, vseg = tid & 3;                 // 64 rows x 4 x 16B
    const unsigned voff  = (unsigned)((vrow * EMB + vseg * 8) * 2);
    const unsigned wlds0 = (unsigned)(uintptr_t)&wbuf[0][vrow * 32 + vseg * 8];
    const unsigned wlds1 = (unsigned)(uintptr_t)&wbuf[1][vrow * 32 + vseg * 8];

    async_b128(wlds0, voff, Wth);                  // prologue: chunk 0
    v8f acc[4] = {};
    for (int kb = 0; kb < EMB; kb += 32) {
        const int cur = (kb >> 5) & 1;
        if (kb + 32 < EMB) {
            async_b128(cur ? wlds0 : wlds1, voff, Wth + kb + 32);
            wait_async1();                         // chunk kb done; next still in flight
        } else {
            wait_async0();
        }
        __syncthreads();
        const v16bf a = load_a_bf16(Xw + kb, EMB, lane);
        const unsigned short* wb = &wbuf[cur][0];
#pragma unroll
        for (int j = 0; j < 4; ++j)
            acc[j] = wmma_bf16(a, load_b_col_bf16(wb + j * 16 * 32, 32, lane), acc[j]);
        __syncthreads();                           // done reading before refill (next iter)
    }

    const size_t planesz = (size_t)BATCH * HEADS * SEQ * HDIM;
    const int col = lane & 15, half = lane >> 4;
    if (which == 2) {
        unsigned short* dstT = qkv + 2 * planesz + ((size_t)b * HEADS + h) * SEQ * HDIM; // [D][T]
#pragma unroll
        for (int r = 0; r < 8; ++r) {
            const int t = row0 + r + 8 * half;
#pragma unroll
            for (int j = 0; j < 4; ++j)
                dstT[(size_t)(j * 16 + col) * SEQ + t] = f2bfu(acc[j][r]);
        }
    } else {
        const float scale = (which == 0) ? 0.125f : 1.0f;
        unsigned short* dst = qkv + (size_t)which * planesz
                            + (((size_t)b * HEADS + h) * SEQ + row0) * HDIM;
#pragma unroll
        for (int r = 0; r < 8; ++r) {
#pragma unroll
            for (int j = 0; j < 4; ++j)
                dst[(size_t)(r + 8 * half) * HDIM + j * 16 + col] = f2bfu(acc[j][r] * scale);
        }
    }
}

// =====================================================================
// Kernel 2: flash attention, 128 query rows per block (16 per wave).
// K (32x64) and V^T (64x32) chunks async-staged to double-buffered LDS;
// wait_asynccnt<=2 keeps the next chunk in flight during compute.
// P bounced through LDS (s_wait_dscnt); all fragments are b128 loads.
// =====================================================================
__global__ __launch_bounds__(256) void attn_kernel(
    const unsigned short* __restrict__ qkv, unsigned short* __restrict__ comb)
{
    __shared__ unsigned short kbuf[2][32 * 64];   // K chunk [s][d]
    __shared__ unsigned short vbuf[2][64 * 32];   // V^T chunk [d][s]
    __shared__ unsigned short pbuf[8][16 * 40];   // per-wave 16x32 P tile

    const int tid  = threadIdx.x;
    const int lane = tid & 31;
    const int wave = tid >> 5;
    const int col = lane & 15, half = lane >> 4;
    const int bh = blockIdx.y;
    const int b = bh / HEADS, h = bh % HEADS;
    const int q0 = blockIdx.x * 128 + wave * 16;

    const size_t planesz = (size_t)BATCH * HEADS * SEQ * HDIM;
    const unsigned short* Q  = qkv + (size_t)bh * SEQ * HDIM + (size_t)q0 * HDIM;
    const unsigned short* K  = qkv + planesz + (size_t)bh * SEQ * HDIM;
    const unsigned short* Vt = qkv + 2 * planesz + (size_t)bh * SEQ * HDIM;   // [D][T]

    const int krow = tid >> 3, kseg = tid & 7;            // K: 32 rows x 8 x 16B
    const int vrow = tid >> 2, vseg = tid & 3;            // V^T: 64 rows x 4 x 16B
    const unsigned kvoff = (unsigned)((krow * HDIM + kseg * 8) * 2);
    const unsigned vvoff = (unsigned)((vrow * SEQ + vseg * 8) * 2);
    const unsigned klds0 = (unsigned)(uintptr_t)&kbuf[0][krow * 64 + kseg * 8];
    const unsigned klds1 = (unsigned)(uintptr_t)&kbuf[1][krow * 64 + kseg * 8];
    const unsigned vlds0 = (unsigned)(uintptr_t)&vbuf[0][vrow * 32 + vseg * 8];
    const unsigned vlds1 = (unsigned)(uintptr_t)&vbuf[1][vrow * 32 + vseg * 8];

    const v16bf aq0 = load_a_bf16(Q,      HDIM, lane);
    const v16bf aq1 = load_a_bf16(Q + 32, HDIM, lane);

    v8f o0 = {}, o1 = {}, o2 = {}, o3 = {};
    float m[8], l[8];
#pragma unroll
    for (int r = 0; r < 8; ++r) { m[r] = -3.0e38f; l[r] = 0.0f; }

    unsigned short* pb = &pbuf[wave][0];

    async_b128(klds0, kvoff, K);                   // prologue: chunk 0
    async_b128(vlds0, vvoff, Vt);

    for (int s0 = 0; s0 < SEQ; s0 += 32) {
        const int cur = (s0 >> 5) & 1;
        if (s0 + 32 < SEQ) {
            async_b128(cur ? klds0 : klds1, kvoff, K + (size_t)(s0 + 32) * HDIM);
            async_b128(cur ? vlds0 : vlds1, vvoff, Vt + s0 + 32);
            wait_async2();                         // this chunk done; next (2 loads) in flight
        } else {
            wait_async0();
        }
        __syncthreads();

        const unsigned short* kb = &kbuf[cur][0];
        const unsigned short* vb = &vbuf[cur][0];

        v8f sc0 = {}, sc1 = {};
        sc0 = wmma_bf16(aq0, load_b_col_bf16(kb,                64, lane), sc0);
        sc0 = wmma_bf16(aq1, load_b_col_bf16(kb + 32,           64, lane), sc0);
        sc1 = wmma_bf16(aq0, load_b_col_bf16(kb + 16 * 64,      64, lane), sc1);
        sc1 = wmma_bf16(aq1, load_b_col_bf16(kb + 16 * 64 + 32, 64, lane), sc1);

#pragma unroll
        for (int r = 0; r < 8; ++r) {
            float mx = fmaxf(sc0[r], sc1[r]);
#pragma unroll
            for (int d = 1; d < 16; d <<= 1) mx = fmaxf(mx, __shfl_xor(mx, d));
            const float mn   = fmaxf(m[r], mx);
            const float corr = __expf(m[r] - mn);
            const float p0 = __expf(sc0[r] - mn);
            const float p1 = __expf(sc1[r] - mn);
            float rs = p0 + p1;
#pragma unroll
            for (int d = 1; d < 16; d <<= 1) rs += __shfl_xor(rs, d);
            l[r] = l[r] * corr + rs;
            m[r] = mn;
            o0[r] *= corr; o1[r] *= corr; o2[r] *= corr; o3[r] *= corr;
            pb[(r + 8 * half) * 40 + col]      = f2bfu(p0);
            pb[(r + 8 * half) * 40 + 16 + col] = f2bfu(p1);
        }
        asm volatile("s_wait_dscnt 0x0" ::: "memory");  // LDS write -> read (same wave)

        const v16bf ap = load_a_bf16(pb, 40, lane);
        o0 = wmma_bf16(ap, load_b_col_bf16(vb,           32, lane), o0);
        o1 = wmma_bf16(ap, load_b_col_bf16(vb + 16 * 32, 32, lane), o1);
        o2 = wmma_bf16(ap, load_b_col_bf16(vb + 32 * 32, 32, lane), o2);
        o3 = wmma_bf16(ap, load_b_col_bf16(vb + 48 * 32, 32, lane), o3);

        __syncthreads();                           // done reading before refill
    }

    unsigned short* out = comb + ((size_t)b * SEQ + q0) * EMB + h * HDIM;
#pragma unroll
    for (int r = 0; r < 8; ++r) {
        const float inv = 1.0f / l[r];
        const size_t off = (size_t)(r + 8 * half) * EMB;
        out[off + col]      = f2bfu(o0[r] * inv);
        out[off + 16 + col] = f2bfu(o1[r] * inv);
        out[off + 32 + col] = f2bfu(o2[r] * inv);
        out[off + 48 + col] = f2bfu(o3[r] * inv);
    }
}

// =====================================================================
// Kernel 3: out = x + combined @ Wo^T. 128x64 tile per block, Wo chunks
// async-staged to double-buffered LDS.
// =====================================================================
__global__ __launch_bounds__(256) void out_proj_kernel(
    const unsigned short* __restrict__ comb, const unsigned short* __restrict__ wob,
    const float* __restrict__ x, float* __restrict__ out)
{
    __shared__ unsigned short wbuf[2][64 * 32];   // Wo chunk [n 64][k 32]

    const int tid = threadIdx.x;
    const int lane = tid & 31;
    const int wave = tid >> 5;
    const int row0 = blockIdx.x * 128 + wave * 16;
    const int colbase = blockIdx.y * 64;
    const int b = blockIdx.z;

    const unsigned short* Cm = comb + ((size_t)b * SEQ + row0) * EMB;

    const int vrow = tid >> 2, vseg = tid & 3;
    const unsigned voff  = (unsigned)((vrow * EMB + vseg * 8) * 2);
    const unsigned wlds0 = (unsigned)(uintptr_t)&wbuf[0][vrow * 32 + vseg * 8];
    const unsigned wlds1 = (unsigned)(uintptr_t)&wbuf[1][vrow * 32 + vseg * 8];
    const unsigned short* Wb = wob + (size_t)colbase * EMB;

    async_b128(wlds0, voff, Wb);                   // prologue: chunk 0
    v8f acc[4] = {};
    for (int kb = 0; kb < EMB; kb += 32) {
        const int cur = (kb >> 5) & 1;
        if (kb + 32 < EMB) {
            async_b128(cur ? wlds0 : wlds1, voff, Wb + kb + 32);
            wait_async1();
        } else {
            wait_async0();
        }
        __syncthreads();
        const v16bf a = load_a_bf16(Cm + kb, EMB, lane);
        const unsigned short* wbp = &wbuf[cur][0];
#pragma unroll
        for (int j = 0; j < 4; ++j)
            acc[j] = wmma_bf16(a, load_b_col_bf16(wbp + j * 16 * 32, 32, lane), acc[j]);
        __syncthreads();
    }

    const int col = lane & 15, half = lane >> 4;
    const float* xr   = x   + ((size_t)b * SEQ + row0) * EMB + colbase;
    float*       orow = out + ((size_t)b * SEQ + row0) * EMB + colbase;
#pragma unroll
    for (int r = 0; r < 8; ++r) {
        const size_t off = (size_t)(r + 8 * half) * EMB;
#pragma unroll
        for (int j = 0; j < 4; ++j)
            orow[off + j * 16 + col] = xr[off + j * 16 + col] + acc[j][r];
    }
}

extern "C" void kernel_launch(void* const* d_in, const int* in_sizes, int n_in,
                              void* d_out, int out_size, void* d_ws, size_t ws_size,
                              hipStream_t stream)
{
    const float* x  = (const float*)d_in[0];
    const float* Wq = (const float*)d_in[1];
    const float* Wk = (const float*)d_in[2];
    const float* Wv = (const float*)d_in[3];
    const float* Wo = (const float*)d_in[4];
    float* out = (float*)d_out;

    // workspace layout (ushort elems): xb | wt[3] | wob | qkv | comb  (~50 MB)
    unsigned short* xb   = (unsigned short*)d_ws;
    unsigned short* wt   = xb  + (size_t)BATCH * SEQ * EMB;
    unsigned short* wob  = wt  + (size_t)3 * HEADS * HDIM * EMB;
    unsigned short* qkv  = wob + (size_t)EMB * EMB;
    unsigned short* comb = qkv + (size_t)3 * BATCH * HEADS * SEQ * HDIM;

    cvt_kernel<<<dim3((BATCH * SEQ * EMB / 8) / 256), 256, 0, stream>>>(x, xb, BATCH * SEQ * EMB / 8);
    cvt_kernel<<<dim3((EMB * EMB / 8) / 256), 256, 0, stream>>>(Wo, wob, EMB * EMB / 8);
    cvtT_kernel<<<dim3((HEADS * EMB * HDIM / 2) / 256), 256, 0, stream>>>(Wq, wt);
    cvtT_kernel<<<dim3((HEADS * EMB * HDIM / 2) / 256), 256, 0, stream>>>(Wk, wt + (size_t)HEADS * HDIM * EMB);
    cvtT_kernel<<<dim3((HEADS * EMB * HDIM / 2) / 256), 256, 0, stream>>>(Wv, wt + (size_t)2 * HEADS * HDIM * EMB);

    proj_qkv_kernel<<<dim3(SEQ / 128, HEADS, BATCH * 3), 256, 0, stream>>>(xb, wt, qkv);
    attn_kernel<<<dim3(SEQ / 128, BATCH * HEADS), 256, 0, stream>>>(qkv, comb);
    out_proj_kernel<<<dim3(SEQ / 128, EMB / 64, BATCH), 256, 0, stream>>>(comb, wob, x, out);
}
